// MHSAWithRelativePosEncoding_84688165143002
// MI455X (gfx1250) — matrix operand order repaired and checked by reading, not compiled, not run on hardware
//
#include <hip/hip_runtime.h>

// ---------------------------------------------------------------------------
// MHSA with relative position encoding (Transformer-XL style), CDNA5/gfx1250.
// All matmuls via v_wmma_f32_16x16x32_bf16 (fp32 accumulate), flash-softmax,
// analytic rel-shift: scores[q,k] = content[q,k] + pos[q, k - q + S - 1].
// Operand tiles double-buffered in LDS via async global->LDS DMA (ASYNCcnt),
// overlapping the copy of tile t+1 with the WMMA work on tile t.
// ---------------------------------------------------------------------------

#define BB 4
#define SS 1024
#define DD 512
#define HH 8
#define HD 64
#define PPAD 2048   // 2S-1 = 2047 positional rows, padded to 2048 with a zero row

typedef __attribute__((ext_vector_type(16))) __bf16 bf16x16;
typedef __attribute__((ext_vector_type(8)))  float  f32x8;

union B16Frag {
    bf16x16 bf;
    uint4   q[2];
    unsigned short u16[16];
};

__device__ __forceinline__ unsigned short f2bf(float f) {
    unsigned int u = __float_as_uint(f);
    unsigned int r = u + 0x7FFFu + ((u >> 16) & 1u);   // RNE
    return (unsigned short)(r >> 16);
}

__device__ __forceinline__ f32x8 wmma_bf16(bf16x16 a, bf16x16 b, f32x8 c) {
    return __builtin_amdgcn_wmma_f32_16x16x32_bf16(
        false, a, false, b, (short)0, c, false, false);
}

// Async 16-byte global -> LDS copy (CDNA5 GLOBAL_LOAD_ASYNC_TO_LDS_B128).
// lds_dst: pointer into __shared__ (flat low 32 bits == LDS offset).
__device__ __forceinline__ void async_ld_b128(void* lds_dst, const void* gsrc) {
    unsigned lo = (unsigned)(unsigned long long)lds_dst;
    asm volatile("global_load_async_to_lds_b128 %0, %1, off"
                 :: "v"(lo), "v"(gsrc) : "memory");
}
__device__ __forceinline__ void async_wait_all() {
    asm volatile("s_wait_asynccnt 0" ::: "memory");
}

// A fragment: 16x32 (MxK) bf16 row-major source.
// lane L: M = L%16, half = L/16. elems 0..7 -> K = k0+half*8..+7 ;
// elems 8..15 -> K = k0+16+half*8..+7   (per ISA 16-bit A layout table)
__device__ __forceinline__ bf16x16 load_frag_a(const unsigned short* base,
                                               int row0, int k0, int ld) {
    int lane = threadIdx.x & 31;
    int m    = row0 + (lane & 15);
    int half = lane >> 4;
    const unsigned short* p = base + (size_t)m * ld + k0 + half * 8;
    B16Frag f;
    f.q[0] = *(const uint4*)p;
    f.q[1] = *(const uint4*)(p + 16);
    return f.bf;
}

// B fragment (32x16 KxN) loaded from an NxK row-major matrix ("@ W^T" operand).
// lane L: N = n0 + L%16 ; elems 0..7 -> K = k0 + (L/16)*8 + 0..7 ; elems 8..15 -> K = k0 + 16 + (L/16)*8 + 0..7
__device__ __forceinline__ bf16x16 load_frag_bt(const unsigned short* base,
                                                int n0, int k0, int ld) {
    int lane = threadIdx.x & 31;
    int n    = n0 + (lane & 15);
    int half = lane >> 4;
    const unsigned short* p = base + (size_t)n * ld + k0 + half * 8;
    B16Frag f;
    f.q[0] = *(const uint4*)p;
    f.q[1] = *(const uint4*)(p + 16);
    return f.bf;
}

// ---------------------------------------------------------------------------
// fp32 -> bf16 conversion kernels
// ---------------------------------------------------------------------------
__global__ __launch_bounds__(256) void cvt_f32_bf16(const float* __restrict__ s,
                                                    unsigned short* __restrict__ d,
                                                    int n) {
    int i = blockIdx.x * 256 + threadIdx.x;
    if (i < n) d[i] = f2bf(s[i]);
}

// pos_emb = pe_matrix rows [ML-S+1 .. ML+S-1] = rows 1..2047 of the 2049-row pe.
// Store 2048 rows, last one zero (padding so pos-GEMM tiles never go OOB).
__global__ __launch_bounds__(256) void cvt_pe(const float* __restrict__ pe,
                                              unsigned short* __restrict__ d) {
    int i = blockIdx.x * 256 + threadIdx.x;   // n = 2048*512
    int r = i >> 9, c = i & 511;
    float v = (r < 2047) ? pe[(size_t)(r + 1) * DD + c] : 0.0f;
    d[i] = f2bf(v);
}

// ---------------------------------------------------------------------------
// Generic C = A @ B^T GEMM, K = 512 fixed. A: Mx512 bf16, B: 512x512 bf16
// (row-major NxK). Block = 128 threads (4 waves), 64x64 tile per block;
// B k-slab double-buffered in LDS via async DMA, shared by all 4 waves.
// MODE 0: Q-proj  -> qu = val+bq+u, qv = val+bq+v, per-head (b,h,s,d)
// MODE 1: K-proj  -> kbuf (b,h,s,d)
// MODE 2: V-proj  -> vT   (b,h,d,s)   (transposed for attn@V B-operand)
// MODE 3: pos-proj-> pkbuf (h,p,d)
// MODE 4: out-proj-> d_out fp32 (+bias)
// ---------------------------------------------------------------------------
template <int MODE>
__global__ __launch_bounds__(128) void gemm_bt(
    const unsigned short* __restrict__ A, const unsigned short* __restrict__ Bm,
    const float* __restrict__ bias, const float* __restrict__ uvec,
    const float* __restrict__ vvec, unsigned short* __restrict__ out0,
    unsigned short* __restrict__ out1, float* __restrict__ outf) {
    __shared__ __attribute__((aligned(16))) unsigned short ldsB[2][64 * 32];

    const int tid = threadIdx.x;
    const int w = tid >> 5, lane = tid & 31;
    const int ln = lane & 15, half = lane >> 4;
    const int m0 = blockIdx.x * 64, n0 = blockIdx.y * 64;

    auto stageB = [&](int buf, int k0) {
        for (int i = 0; i < 2; i++) {
            int c = tid + 128 * i;              // 16B chunk id, 0..255
            int row = c >> 2, off = (c & 3) * 8;
            async_ld_b128(ldsB[buf] + row * 32 + off,
                          Bm + (size_t)(n0 + row) * DD + k0 + off);
        }
    };

    f32x8 acc[4] = {};

    stageB(0, 0);
    for (int k0 = 0; k0 < DD; k0 += 32) {
        const int cur = (k0 >> 5) & 1;
        async_wait_all();
        __syncthreads();
        if (k0 + 32 < DD) stageB(cur ^ 1, k0 + 32);   // prefetch next slab

        bf16x16 a0 = load_frag_a(A, m0 + w * 16, k0, DD);
        for (int ns = 0; ns < 4; ns++) {
            bf16x16 b = load_frag_bt(ldsB[cur], ns * 16, 0, 32);
            acc[ns] = wmma_bf16(a0, b, acc[ns]);
        }
    }

    for (int ns = 0; ns < 4; ns++) {
        for (int r = 0; r < 8; r++) {
            int m = m0 + w * 16 + r + 8 * half;
            int n = n0 + ns * 16 + ln;
            float val = acc[ns][r];
            if (MODE == 0) {
                val += bias[n];
                int b = m >> 10, s = m & 1023, h = n >> 6, d = n & 63;
                size_t o = (((size_t)(b * HH + h) * SS) + s) * HD + d;
                out0[o] = f2bf(val + uvec[n]);
                out1[o] = f2bf(val + vvec[n]);
            } else if (MODE == 1) {
                int b = m >> 10, s = m & 1023, h = n >> 6, d = n & 63;
                out0[(((size_t)(b * HH + h) * SS) + s) * HD + d] = f2bf(val + bias[n]);
            } else if (MODE == 2) {
                int b = m >> 10, s = m & 1023, h = n >> 6, d = n & 63;
                out0[(((size_t)(b * HH + h) * HD) + d) * SS + s] = f2bf(val + bias[n]);
            } else if (MODE == 3) {
                int h = n >> 6, d = n & 63;
                out0[((size_t)h * PPAD + m) * HD + d] = f2bf(val);
            } else {
                outf[(size_t)m * DD + n] = val + bias[n];
            }
        }
    }
}

// ---------------------------------------------------------------------------
// Fused flash attention with relative positions.
// Block = 128 threads = 4 waves, q rows [q0blk, q0blk+64). All waves iterate
// the same key-tiles; the K-tile (8KB), V^T-tile (8KB) and the union of the 4
// waves' positional bands (128 rows of pk, 16KB) are double-buffered in LDS
// via async DMA (prefetch of tile t+1 overlaps WMMA work on tile t).
// Per key-tile per wave: 8 wmma content + 10 wmma positional + 8 wmma attn@V
// + online softmax.
// ---------------------------------------------------------------------------
__global__ __launch_bounds__(128) void attn_rel_flash(
    const unsigned short* __restrict__ qu, const unsigned short* __restrict__ qv,
    const unsigned short* __restrict__ kbuf, const unsigned short* __restrict__ vT,
    const unsigned short* __restrict__ pk, unsigned short* __restrict__ attnout) {
    __shared__ __attribute__((aligned(16))) unsigned short ldsK[2][64 * 64];   // key x feat
    __shared__ __attribute__((aligned(16))) unsigned short ldsV[2][64 * 64];   // feat x key
    __shared__ __attribute__((aligned(16))) unsigned short ldsP[2][128 * 64];  // p x feat
    __shared__ __attribute__((aligned(16))) float posT[4][16 * 80];
    __shared__ __attribute__((aligned(16))) unsigned short pT[4][16 * 64];

    const int tid = threadIdx.x;
    const int w = tid >> 5, lane = tid & 31;
    const int ln = lane & 15, half = lane >> 4;
    const int bh = blockIdx.y;                 // b*H + h
    const int h  = bh & (HH - 1);
    const int b  = bh >> 3;
    const int q0blk = blockIdx.x * 64;
    const int q0 = q0blk + w * 16;

    const unsigned short* quh = qu   + (size_t)bh * SS * HD;
    const unsigned short* qvh = qv   + (size_t)bh * SS * HD;
    const unsigned short* kh  = kbuf + (size_t)bh * SS * HD;
    const unsigned short* vh  = vT   + (size_t)bh * HD * SS;
    const unsigned short* pkh = pk   + (size_t)h * PPAD * HD;

    auto stage_tile = [&](int buf, int k0) {
        const int pb0 = k0 - q0blk + (SS - 64);       // base row of shared pk slab
        for (int i = 0; i < 4; i++) {                 // K tile: contiguous 8KB
            int c = tid + 128 * i;
            async_ld_b128(ldsK[buf] + c * 8, kh + (size_t)k0 * HD + c * 8);
        }
        for (int i = 0; i < 4; i++) {                 // V^T tile: 64 rows, stride S
            int c = tid + 128 * i;
            int row = c >> 3, off = (c & 7) * 8;
            async_ld_b128(ldsV[buf] + row * 64 + off,
                          vh + (size_t)row * SS + k0 + off);
        }
        for (int i = 0; i < 8; i++) {                 // pk band: contiguous 16KB
            int c = tid + 128 * i;
            async_ld_b128(ldsP[buf] + c * 8, pkh + (size_t)pb0 * HD + c * 8);
        }
    };

    bf16x16 quA[2], qvA[2];
    for (int ks = 0; ks < 2; ks++) {
        quA[ks] = load_frag_a(quh, q0, ks * 32, HD);
        qvA[ks] = load_frag_a(qvh, q0, ks * 32, HD);
    }

    f32x8 accO[4] = {};
    float mrun[8], lrun[8];
    for (int r = 0; r < 8; r++) { mrun[r] = -__builtin_inff(); lrun[r] = 0.0f; }

    float*          myPos = posT[w];
    unsigned short* myP   = pT[w];
    const int pw = 48 - 16 * w;   // wave's band offset inside the shared pk slab

    stage_tile(0, 0);
    for (int kt = 0; kt < SS / 64; kt++) {
        const int k0  = kt * 64;
        const int cur = kt & 1;

        async_wait_all();
        __syncthreads();
        if (kt + 1 < SS / 64) stage_tile(cur ^ 1, k0 + 64);   // overlap DMA

        // ---- content scores: (q+u) @ k^T, 16x64 tile ----
        f32x8 cs[4] = {};
        for (int ks = 0; ks < 2; ks++)
            for (int ns = 0; ns < 4; ns++)
                cs[ns] = wmma_bf16(quA[ks],
                                   load_frag_bt(ldsK[cur], ns * 16, ks * 32, HD),
                                   cs[ns]);

        // ---- positional band: 16x80 Toeplitz slice of (q+v) @ pk^T ----
        for (int ns = 0; ns < 5; ns++) {
            f32x8 pp = {};
            for (int ks = 0; ks < 2; ks++)
                pp = wmma_bf16(qvA[ks],
                               load_frag_bt(ldsP[cur], pw + ns * 16, ks * 32, HD),
                               pp);
            for (int r = 0; r < 8; r++)
                myPos[(r + 8 * half) * 80 + ns * 16 + ln] = pp[r];
        }
        asm volatile("s_wait_dscnt 0" ::: "memory");

        // ---- scores + online softmax (fused, scores overwrite cs in place) --
        for (int r = 0; r < 8; r++) {
            int m = r + 8 * half;
            float mx = -__builtin_inff();
            for (int ns = 0; ns < 4; ns++) {
                int j = ns * 16 + ln - m + 15;             // rel-shift gather
                float s = (cs[ns][r] + myPos[m * 80 + j]) * 0.125f;
                cs[ns][r] = s;
                mx = fmaxf(mx, s);
            }
            for (int d = 1; d < 16; d <<= 1)
                mx = fmaxf(mx, __shfl_xor(mx, d, 32));
            float newm  = fmaxf(mrun[r], mx);
            float scale = __expf(mrun[r] - newm);
            mrun[r] = newm;
            float rs = 0.0f;
            for (int ns = 0; ns < 4; ns++) {
                float p = __expf(cs[ns][r] - newm);
                rs += p;
                myP[m * 64 + ns * 16 + ln] = f2bf(p);
            }
            for (int d = 1; d < 16; d <<= 1)
                rs += __shfl_xor(rs, d, 32);
            lrun[r] = lrun[r] * scale + rs;
            for (int ds = 0; ds < 4; ds++) accO[ds][r] *= scale;
        }
        asm volatile("s_wait_dscnt 0" ::: "memory");

        // ---- attn @ V : P-tile (C-layout -> A-layout via LDS), V^T in LDS ----
        bf16x16 pa0 = load_frag_a(myP, 0, 0, 64);
        bf16x16 pa1 = load_frag_a(myP, 0, 32, 64);
        for (int ds = 0; ds < 4; ds++) {
            accO[ds] = wmma_bf16(pa0, load_frag_bt(ldsV[cur], ds * 16, 0, 64), accO[ds]);
            accO[ds] = wmma_bf16(pa1, load_frag_bt(ldsV[cur], ds * 16, 32, 64), accO[ds]);
        }
    }

    // ---- finalize and scatter to (B*S, D) bf16 for the output projection ----
    for (int ds = 0; ds < 4; ds++)
        for (int r = 0; r < 8; r++) {
            int m = r + 8 * half;
            float o = accO[ds][r] / lrun[r];
            size_t row = (size_t)b * SS + q0 + m;
            attnout[row * DD + h * HD + ds * 16 + ln] = f2bf(o);
        }
}

// ---------------------------------------------------------------------------
extern "C" void kernel_launch(void* const* d_in, const int* in_sizes, int n_in,
                              void* d_out, int out_size, void* d_ws, size_t ws_size,
                              hipStream_t stream) {
    const float* x  = (const float*)d_in[0];
    const float* Wq = (const float*)d_in[1];
    const float* bq = (const float*)d_in[2];
    const float* Wk = (const float*)d_in[3];
    const float* bk = (const float*)d_in[4];
    const float* Wv = (const float*)d_in[5];
    const float* bv = (const float*)d_in[6];
    const float* Wp = (const float*)d_in[7];
    const float* uu = (const float*)d_in[8];
    const float* vv = (const float*)d_in[9];
    const float* Wo = (const float*)d_in[10];
    const float* bo = (const float*)d_in[11];
    const float* pe = (const float*)d_in[12];
    float* out = (float*)d_out;

    char* ws = (char*)d_ws;
    unsigned short* xbf  = (unsigned short*)ws;  ws += (size_t)BB * SS * DD * 2;
    unsigned short* wqb  = (unsigned short*)ws;  ws += (size_t)DD * DD * 2;
    unsigned short* wkb  = (unsigned short*)ws;  ws += (size_t)DD * DD * 2;
    unsigned short* wvb  = (unsigned short*)ws;  ws += (size_t)DD * DD * 2;
    unsigned short* wpb  = (unsigned short*)ws;  ws += (size_t)DD * DD * 2;
    unsigned short* wob  = (unsigned short*)ws;  ws += (size_t)DD * DD * 2;
    unsigned short* pebf = (unsigned short*)ws;  ws += (size_t)PPAD * DD * 2;
    unsigned short* qub  = (unsigned short*)ws;  ws += (size_t)BB * HH * SS * HD * 2;
    unsigned short* qvb  = (unsigned short*)ws;  ws += (size_t)BB * HH * SS * HD * 2;
    unsigned short* kb   = (unsigned short*)ws;  ws += (size_t)BB * HH * SS * HD * 2;
    unsigned short* vtb  = (unsigned short*)ws;  ws += (size_t)BB * HH * HD * SS * 2;
    unsigned short* pkb  = (unsigned short*)ws;  ws += (size_t)HH * PPAD * HD * 2;
    unsigned short* aob  = (unsigned short*)ws;  ws += (size_t)BB * SS * DD * 2;

    // 1) convert inputs to bf16
    cvt_f32_bf16<<<(BB * SS * DD + 255) / 256, 256, 0, stream>>>(x, xbf, BB * SS * DD);
    cvt_f32_bf16<<<(DD * DD + 255) / 256, 256, 0, stream>>>(Wq, wqb, DD * DD);
    cvt_f32_bf16<<<(DD * DD + 255) / 256, 256, 0, stream>>>(Wk, wkb, DD * DD);
    cvt_f32_bf16<<<(DD * DD + 255) / 256, 256, 0, stream>>>(Wv, wvb, DD * DD);
    cvt_f32_bf16<<<(DD * DD + 255) / 256, 256, 0, stream>>>(Wp, wpb, DD * DD);
    cvt_f32_bf16<<<(DD * DD + 255) / 256, 256, 0, stream>>>(Wo, wob, DD * DD);
    cvt_pe<<<(PPAD * DD + 255) / 256, 256, 0, stream>>>(pe, pebf);

    // 2) projections (WMMA GEMMs)
    dim3 gqkv(BB * SS / 64, DD / 64), blk(128);
    gemm_bt<0><<<gqkv, blk, 0, stream>>>(xbf, wqb, bq, uu, vv, qub, qvb, nullptr);
    gemm_bt<1><<<gqkv, blk, 0, stream>>>(xbf, wkb, bk, nullptr, nullptr, kb, nullptr, nullptr);
    gemm_bt<2><<<gqkv, blk, 0, stream>>>(xbf, wvb, bv, nullptr, nullptr, vtb, nullptr, nullptr);
    dim3 gpk(PPAD / 64, DD / 64);
    gemm_bt<3><<<gpk, blk, 0, stream>>>(pebf, wpb, nullptr, nullptr, nullptr, pkb, nullptr, nullptr);

    // 3) fused flash attention with relative positions
    dim3 ga(SS / 64, BB * HH);
    attn_rel_flash<<<ga, blk, 0, stream>>>(qub, qvb, kb, vtb, pkb, aob);

    // 4) output projection (fp32 result)
    gemm_bt<4><<<gqkv, blk, 0, stream>>>(aob, wob, bo, nullptr, nullptr, nullptr, nullptr, out);
}
